// QuantLinear_23983097381438
// MI455X (gfx1250) — matrix operand order repaired and checked by reading, not compiled, not run on hardware
//
#include <hip/hip_runtime.h>

// GPTQ 4-bit dequant + GEMM fused kernel for gfx1250 (MI455X).
// out[M,N] = x[M,K] @ dequant(qweight,qzeros,scales,g_idx)[K,N] + bias
// M = 8192, K = 4096, N = 4096, GROUP_SIZE = 128, 4-bit pack = 8/int32.
//
// Double-buffered LDS pipeline, one barrier per K-tile.
// A-tile staged via GLOBAL_LOAD_ASYNC_TO_LDS_B128 (ASYNCcnt path).

typedef __attribute__((ext_vector_type(16))) _Float16 v16h;
typedef __attribute__((ext_vector_type(8)))  _Float16 v8h;
typedef __attribute__((ext_vector_type(8)))  float    v8f;
typedef __attribute__((ext_vector_type(4)))  int      v4i;

#define K_DIM   4096
#define N_DIM   4096
#define NPACK   (N_DIM / 8)       // int32 words per qzeros row
#define TILE_K  64
#define LDSTR   72                // padded LDS row stride in halves (144B)
#define NTILES  (K_DIM / TILE_K)  // 64

#if __has_builtin(__builtin_amdgcn_global_load_async_to_lds_b128)
#define HAVE_ASYNC_LDS 1
#else
#define HAVE_ASYNC_LDS 0
#endif

__device__ __forceinline__ void async_copy_b128(const _Float16* g, _Float16* l) {
#if HAVE_ASYNC_LDS
    typedef __attribute__((address_space(1))) v4i* gas_p;   // global v4i*
    typedef __attribute__((address_space(3))) v4i* las_p;   // LDS v4i*
    __builtin_amdgcn_global_load_async_to_lds_b128(
        (gas_p)(g), (las_p)(l), /*offset=*/0, /*cpol=*/0);
#else
    *(v8h*)l = *(const v8h*)g;
#endif
}

__device__ __forceinline__ void wait_async_then_barrier() {
#if HAVE_ASYNC_LDS
#if __has_builtin(__builtin_amdgcn_s_wait_asynccnt)
    __builtin_amdgcn_s_wait_asynccnt(0);
#else
    asm volatile("s_wait_asynccnt 0x0" ::: "memory");
#endif
#endif
    __syncthreads();
}

__global__ __launch_bounds__(256)
void gptq_wmma_f16_kernel(const _Float16* __restrict__ x,
                          const int*      __restrict__ qweight,
                          const int*      __restrict__ qzeros,
                          const _Float16* __restrict__ scales,
                          const int*      __restrict__ g_idx,
                          const _Float16* __restrict__ bias,
                          _Float16*       __restrict__ out)
{
    __shared__ _Float16 As[2][128 * LDSTR];   // x tile, M-major  [m][k]
    __shared__ _Float16 Bt[2][128 * LDSTR];   // W tile, N-major  [n][k]

    const int tid  = threadIdx.x;
    const int lane = tid & 31;
    const int wave = tid >> 5;          // 0..7
    const int wm   = wave & 3;          // wave M quadrant (32 rows each)
    const int wn   = wave >> 2;         // wave N half (64 cols each)

    const int blockN = blockIdx.x * 128;
    const int blockM = blockIdx.y * 128;

    // A-tile staging: thread -> (row, 64B half-row)
    const int a_row = tid >> 1;             // 0..127
    const int a_col = (tid & 1) * 32;       // in halves

    // B-tile dequant: thread -> (n, 4 qweight rows)
    const int b_n    = tid & 127;           // 0..127
    const int b_rblk = tid >> 7;            // 0..1

    const int n_lane = lane & 15;           // sub-tile index within 16
    const int hi16   = lane >> 4;           // lane-group select

    v8f acc[2][4] = {};                     // zero-init accumulators

    // ---- per-thread staging helpers ----
    auto stage_A = [&](int kt, int buf) {
        const _Float16* xrow = x + (size_t)(blockM + a_row) * K_DIM + kt + a_col;
        _Float16* dst = &As[buf][a_row * LDSTR + a_col];
        #pragma unroll
        for (int c = 0; c < 4; ++c)
            async_copy_b128(xrow + c * 8, dst + c * 8);
    };

    int   wq[4];        // packed 4-bit weights held across the WMMA block
    int   z_cur;
    float sc_cur;
    auto load_B = [&](int kt) {             // global fetch only (issued early)
        const int g  = g_idx[kt];           // uniform per K-tile
        const int gn = blockN + b_n;
        const int zw = qzeros[g * NPACK + (gn >> 3)];
        z_cur  = (zw >> ((gn & 7) * 4)) & 0xF;
        sc_cur = (float)scales[g * N_DIM + gn];
        const int kr8 = kt >> 3;
        #pragma unroll
        for (int i = 0; i < 4; ++i)
            wq[i] = qweight[(size_t)(kr8 + b_rblk * 4 + i) * N_DIM + gn];
    };
    auto store_B = [&](int buf) {           // unpack math + LDS store (late)
        #pragma unroll
        for (int i = 0; i < 4; ++i) {
            const int r = b_rblk * 4 + i;   // 0..7 (K/8 rows in tile)
            v8h hv;
            #pragma unroll
            for (int j = 0; j < 8; ++j) {
                const int q = (wq[i] >> (4 * j)) & 0xF;
                hv[j] = (_Float16)((float)(q - z_cur) * sc_cur);
            }
            *(v8h*)(&Bt[buf][b_n * LDSTR + r * 8]) = hv;
        }
    };

    auto compute = [&](int buf) {
        #pragma unroll
        for (int ks = 0; ks < TILE_K; ks += 32) {
            // A frags: contiguous-K 16-bit A 16x32 layout (same k order as B)
            //   lanes 0-15  : M=lane,    K = {0..15}
            //   lanes 16-31 : M=lane-16, K = {16..31}
            v16h afrag[2];
            #pragma unroll
            for (int i = 0; i < 2; ++i) {
                const int m = wm * 32 + i * 16 + n_lane;
                const _Float16* ap = &As[buf][m * LDSTR + ks + hi16 * 16];
                v8h lo = *(const v8h*)(ap);
                v8h hi = *(const v8h*)(ap + 8);
                afrag[i] = __builtin_shufflevector(lo, hi,
                    0,1,2,3,4,5,6,7,8,9,10,11,12,13,14,15);
            }
            // B frags: lane = column N, 16 contiguous K halves
            v16h bfrag[4];
            #pragma unroll
            for (int j = 0; j < 4; ++j) {
                const int n = wn * 64 + j * 16 + n_lane;
                bfrag[j] = *(const v16h*)(&Bt[buf][n * LDSTR + ks + hi16 * 16]);
            }
            #pragma unroll
            for (int i = 0; i < 2; ++i)
                #pragma unroll
                for (int j = 0; j < 4; ++j)
                    acc[i][j] = __builtin_amdgcn_wmma_f32_16x16x32_f16(
                        false, afrag[i], false, bfrag[j],
                        (short)0, acc[i][j], false, false);
        }
    };

    // ---- software pipeline: prologue ----
    stage_A(0, 0);
    load_B(0);
    store_B(0);
    wait_async_then_barrier();

    // ---- main loop: compute buf, fill buf^1, one barrier/iter ----
    for (int t = 0; t < NTILES; ++t) {
        const int buf = t & 1;
        if (t + 1 < NTILES) {
            const int ktn = (t + 1) * TILE_K;
            stage_A(ktn, buf ^ 1);          // async DMA overlaps compute
            load_B(ktn);                    // global fetch overlaps compute
        }
        compute(buf);
        if (t + 1 < NTILES)
            store_B(buf ^ 1);               // VALU unpack co-executes w/ XDL
        wait_async_then_barrier();
    }

    // ---- epilogue: f32 -> f16, + bias, store ----
    // C/D layout: VGPR r, lanes 0-15 -> M=r, N=lane; lanes 16-31 -> M=8+r
    #pragma unroll
    for (int j = 0; j < 4; ++j) {
        const int gn = blockN + wn * 64 + j * 16 + n_lane;
        const _Float16 bj = bias[gn];
        #pragma unroll
        for (int i = 0; i < 2; ++i) {
            const int gm0 = blockM + wm * 32 + i * 16 + hi16 * 8;
            #pragma unroll
            for (int r = 0; r < 8; ++r) {
                const _Float16 h = (_Float16)acc[i][j][r] + bj;
                out[(size_t)(gm0 + r) * N_DIM + gn] = h;
            }
        }
    }
}

extern "C" void kernel_launch(void* const* d_in, const int* in_sizes, int n_in,
                              void* d_out, int out_size, void* d_ws, size_t ws_size,
                              hipStream_t stream) {
    (void)in_sizes; (void)n_in; (void)out_size; (void)d_ws; (void)ws_size;
    const _Float16* x       = (const _Float16*)d_in[0];
    const int*      qweight = (const int*)d_in[1];
    const int*      qzeros  = (const int*)d_in[2];
    const _Float16* scales  = (const _Float16*)d_in[3];
    const int*      g_idx   = (const int*)d_in[4];
    const _Float16* bias    = (const _Float16*)d_in[5];
    _Float16*       out     = (_Float16*)d_out;

    // M = 4*2048 = 8192 rows, N = 4096 cols; 128x128 block tiles.
    dim3 grid(N_DIM / 128, 8192 / 128);   // (32, 64)
    dim3 block(256);
    gptq_wmma_f16_kernel<<<grid, block, 0, stream>>>(
        x, qweight, qzeros, scales, g_idx, bias, out);
}
